// GroupQuantizedLinear_11338713661740
// MI455X (gfx1250) — matrix operand order repaired and checked by reading, not compiled, not run on hardware
//
#include <hip/hip_runtime.h>
#include <cstdint>

typedef __attribute__((ext_vector_type(16))) _Float16 v16h;
typedef __attribute__((ext_vector_type(8)))  _Float16 v8h;
typedef __attribute__((ext_vector_type(4)))  _Float16 v4h;
typedef __attribute__((ext_vector_type(8)))  float    v8f;

#define TOKENS 2048
#define NOUT   28672
#define KDIM   4096
#define SPLIT  14336
#define NGROUP 32

#define M_TILE 128
#define N_TILE 128
#define KT     32
#define NITER  (KDIM / KT)   // 128 k-steps

// ---------------------------------------------------------------------------
// CDNA5 async copy to LDS (ASYNCcnt path). ldst: generic pointer to __shared__
// (low 32 bits = wave-relative LDS byte address per ISA flat-aperture rules);
// gsrc: 64-bit global address in a VGPR pair.
// ---------------------------------------------------------------------------
#define ASYNC_B128(ldst, gsrc)                                                  \
    asm volatile("global_load_async_to_lds_b128 %0, %1, off"                    \
                 :: "v"((unsigned)(uintptr_t)(ldst)),                           \
                    "v"((unsigned long long)(uintptr_t)(gsrc))                  \
                 : "memory")

#define WAIT_ASYNC0() asm volatile("s_wait_asynccnt 0x0" ::: "memory")

// ===========================================================================
// Pre-pass 1: dequantize W (round(clip(w,-8,7)) * group scale) -> f16
// 8 contiguous elements/thread (always within one row and one 128-group).
// ===========================================================================
__global__ __launch_bounds__(256)
void dequant_w_kernel(const float* __restrict__ w,
                      const float* __restrict__ scales,
                      _Float16* __restrict__ wq)
{
    const size_t base = ((size_t)blockIdx.x * 256 + threadIdx.x) * 8;
    const int row = (int)(base >> 12);      // /4096
    const int col = (int)(base & 4095);
    const float s = scales[(size_t)row * NGROUP + (col >> 7)];
    const float4 a = *(const float4*)(w + base);
    const float4 b = *(const float4*)(w + base + 4);
    v8h h;
    h[0] = (_Float16)(rintf(fminf(fmaxf(a.x, -8.f), 7.f)) * s);
    h[1] = (_Float16)(rintf(fminf(fmaxf(a.y, -8.f), 7.f)) * s);
    h[2] = (_Float16)(rintf(fminf(fmaxf(a.z, -8.f), 7.f)) * s);
    h[3] = (_Float16)(rintf(fminf(fmaxf(a.w, -8.f), 7.f)) * s);
    h[4] = (_Float16)(rintf(fminf(fmaxf(b.x, -8.f), 7.f)) * s);
    h[5] = (_Float16)(rintf(fminf(fmaxf(b.y, -8.f), 7.f)) * s);
    h[6] = (_Float16)(rintf(fminf(fmaxf(b.z, -8.f), 7.f)) * s);
    h[7] = (_Float16)(rintf(fminf(fmaxf(b.w, -8.f), 7.f)) * s);
    *(v8h*)(wq + base) = h;
}

// ===========================================================================
// Pre-pass 2: x f32 -> f16
// ===========================================================================
__global__ __launch_bounds__(256)
void cvt_x_kernel(const float* __restrict__ x, _Float16* __restrict__ xh)
{
    const size_t base = ((size_t)blockIdx.x * 256 + threadIdx.x) * 8;
    const float4 a = *(const float4*)(x + base);
    const float4 b = *(const float4*)(x + base + 4);
    v8h h;
    h[0] = (_Float16)a.x; h[1] = (_Float16)a.y; h[2] = (_Float16)a.z; h[3] = (_Float16)a.w;
    h[4] = (_Float16)b.x; h[5] = (_Float16)b.y; h[6] = (_Float16)b.z; h[7] = (_Float16)b.w;
    *(v8h*)(xh + base) = h;
}

// ===========================================================================
// Main f16 GEMM: async-to-LDS double buffer, 128x128x32 tile, 8 wave32s (4Mx2N),
// 8x v_wmma_f32_16x16x32_f16 per wave per k-step.
// ===========================================================================
__global__ __launch_bounds__(256)
void gemm_f16_async_kernel(const _Float16* __restrict__ xh,
                           const _Float16* __restrict__ wq,
                           float* __restrict__ out)
{
    __shared__ _Float16 lds_a[2][M_TILE * KT];   // [m][k]
    __shared__ _Float16 lds_b[2][N_TILE * KT];   // [n][k]

    const int tid  = threadIdx.x;
    const int lane = tid & 31;
    const int wid  = tid >> 5;
    const int wm   = wid & 3;    // 4 waves over M: 32 rows each
    const int wn   = wid >> 2;   // 2 waves over N: 64 cols each

    const int M0 = blockIdx.x * M_TILE;   // M-fastest grid: weight stripe L2 reuse
    const int N0 = blockIdx.y * N_TILE;

    v8f acc[2][4];
    {
        const v8f vz = {};
        #pragma unroll
        for (int i = 0; i < 2; ++i)
            #pragma unroll
            for (int j = 0; j < 4; ++j)
                acc[i][j] = vz;
    }

    // tile = 128 rows x 32 halfs = 512 b128 segments; 2 per thread per matrix
    auto issue_async = [&](int k0, int buf) {
        #pragma unroll
        for (int j = 0; j < 2; ++j) {
            const int seg = j * 256 + tid;
            const int row = seg >> 2;        // 4 segments per 64B row
            const int ko  = (seg & 3) << 3;  // half offset within row
            ASYNC_B128(&lds_a[buf][row * KT + ko],
                       xh + (size_t)(M0 + row) * KDIM + k0 + ko);
            ASYNC_B128(&lds_b[buf][row * KT + ko],
                       wq + (size_t)(N0 + row) * KDIM + k0 + ko);
        }
    };

    auto compute = [&](int buf) {
        const int r   = lane & 15;
        const int hi  = lane >> 4;
        const int kbA = hi << 4;   // A: lanes<16 hold K 0..15; lanes>=16 K 16..31
        const int kbB = hi << 4;   // B: lanes<16 hold K 0..15; lanes>=16 K 16..31
        v16h af[2], bf[4];
        #pragma unroll
        for (int i = 0; i < 2; ++i) {
            const _Float16* p = &lds_a[buf][(size_t)(wm * 32 + i * 16 + r) * KT];
            union { v16h v; v8h h[2]; } u;
            u.h[0] = *(const v8h*)(p + kbA);
            u.h[1] = *(const v8h*)(p + 8 + kbA);
            af[i] = u.v;
        }
        #pragma unroll
        for (int j = 0; j < 4; ++j)
            bf[j] = *(const v16h*)(&lds_b[buf][(size_t)(wn * 64 + j * 16 + r) * KT + kbB]);
        #pragma unroll
        for (int i = 0; i < 2; ++i)
            #pragma unroll
            for (int j = 0; j < 4; ++j)
                acc[i][j] = __builtin_amdgcn_wmma_f32_16x16x32_f16(
                    false, af[i], false, bf[j], (short)0, acc[i][j], false, false);
    };

    // pipeline: wait(own batch) -> barrier -> issue next batch -> compute.
    // Copies for k-step it+1 overlap the WMMAs of k-step it.
    issue_async(0, 0);
    for (int it = 0; it < NITER; ++it) {
        const int cur = it & 1;
        WAIT_ASYNC0();
        __syncthreads();
        if (it + 1 < NITER) issue_async((it + 1) * KT, cur ^ 1);
        compute(cur);
    }

    // epilogue: lane = one N column, 8 M rows per v8f accumulator.
    const int ncol = N0 + wn * 64 + (lane & 15);
    #pragma unroll
    for (int i = 0; i < 2; ++i) {
        const int mbase = M0 + wm * 32 + i * 16 + ((lane >> 4) << 3);
        #pragma unroll
        for (int j = 0; j < 4; ++j) {
            const int o = ncol + j * 16;
            const size_t obase = (o < SPLIT)
                ? (size_t)o
                : ((size_t)TOKENS * SPLIT + (size_t)(o - SPLIT));
            #pragma unroll
            for (int rr = 0; rr < 8; ++rr)
                out[obase + (size_t)(mbase + rr) * SPLIT] = acc[i][j][rr];
        }
    }
}

// ===========================================================================
// Fallback (small workspace): fused dequant + GEMM.
// ===========================================================================
__global__ __launch_bounds__(256)
void gql_wmma_fused_kernel(const float* __restrict__ x,
                           const float* __restrict__ w,
                           const float* __restrict__ scales,
                           float* __restrict__ out)
{
    __shared__ _Float16 lds_a[2][M_TILE * KT];
    __shared__ _Float16 lds_b[2][N_TILE * KT];

    const int tid  = threadIdx.x;
    const int lane = tid & 31;
    const int wid  = tid >> 5;
    const int wm   = wid & 3;
    const int wn   = wid >> 2;

    const int M0 = blockIdx.x * M_TILE;
    const int N0 = blockIdx.y * N_TILE;

    v8f acc[2][4];
    {
        const v8f vz = {};
        #pragma unroll
        for (int i = 0; i < 2; ++i)
            #pragma unroll
            for (int j = 0; j < 4; ++j)
                acc[i][j] = vz;
    }

    float4 ar[4], br[4];
    float  sc[4];

    auto load_tiles = [&](int k0) {
        const int g = k0 >> 7;
        #pragma unroll
        for (int j = 0; j < 4; ++j) {
            const int f   = j * 256 + tid;
            const int row = f >> 3;
            const int kk  = (f & 7) << 2;
            ar[j] = *(const float4*)(x + (size_t)(M0 + row) * KDIM + k0 + kk);
            br[j] = *(const float4*)(w + (size_t)(N0 + row) * KDIM + k0 + kk);
            sc[j] = scales[(size_t)(N0 + row) * NGROUP + g];
        }
    };

    auto store_tiles = [&](int buf) {
        #pragma unroll
        for (int j = 0; j < 4; ++j) {
            const int f   = j * 256 + tid;
            const int row = f >> 3;
            const int kk  = (f & 7) << 2;
            v4h ha;
            ha.x = (_Float16)ar[j].x; ha.y = (_Float16)ar[j].y;
            ha.z = (_Float16)ar[j].z; ha.w = (_Float16)ar[j].w;
            *(v4h*)(&lds_a[buf][row * KT + kk]) = ha;
            const float s = sc[j];
            v4h hb;
            hb.x = (_Float16)(rintf(fminf(fmaxf(br[j].x, -8.f), 7.f)) * s);
            hb.y = (_Float16)(rintf(fminf(fmaxf(br[j].y, -8.f), 7.f)) * s);
            hb.z = (_Float16)(rintf(fminf(fmaxf(br[j].z, -8.f), 7.f)) * s);
            hb.w = (_Float16)(rintf(fminf(fmaxf(br[j].w, -8.f), 7.f)) * s);
            *(v4h*)(&lds_b[buf][row * KT + kk]) = hb;
        }
    };

    auto compute = [&](int buf) {
        const int r   = lane & 15;
        const int hi  = lane >> 4;
        const int kbA = hi << 4;
        const int kbB = hi << 4;
        v16h af[2], bf[4];
        #pragma unroll
        for (int i = 0; i < 2; ++i) {
            const _Float16* p = &lds_a[buf][(size_t)(wm * 32 + i * 16 + r) * KT];
            union { v16h v; v8h h[2]; } u;
            u.h[0] = *(const v8h*)(p + kbA);
            u.h[1] = *(const v8h*)(p + 8 + kbA);
            af[i] = u.v;
        }
        #pragma unroll
        for (int j = 0; j < 4; ++j)
            bf[j] = *(const v16h*)(&lds_b[buf][(size_t)(wn * 64 + j * 16 + r) * KT + kbB]);
        #pragma unroll
        for (int i = 0; i < 2; ++i)
            #pragma unroll
            for (int j = 0; j < 4; ++j)
                acc[i][j] = __builtin_amdgcn_wmma_f32_16x16x32_f16(
                    false, af[i], false, bf[j], (short)0, acc[i][j], false, false);
    };

    load_tiles(0);
    store_tiles(0);
    __syncthreads();

    for (int it = 0; it < NITER; ++it) {
        const int cur = it & 1;
        if (it + 1 < NITER) load_tiles((it + 1) * KT);
        compute(cur);
        if (it + 1 < NITER) store_tiles(cur ^ 1);
        __syncthreads();
    }

    const int ncol = N0 + wn * 64 + (lane & 15);
    #pragma unroll
    for (int i = 0; i < 2; ++i) {
        const int mbase = M0 + wm * 32 + i * 16 + ((lane >> 4) << 3);
        #pragma unroll
        for (int j = 0; j < 4; ++j) {
            const int o = ncol + j * 16;
            const size_t obase = (o < SPLIT)
                ? (size_t)o
                : ((size_t)TOKENS * SPLIT + (size_t)(o - SPLIT));
            #pragma unroll
            for (int rr = 0; rr < 8; ++rr)
                out[obase + (size_t)(mbase + rr) * SPLIT] = acc[i][j][rr];
        }
    }
}

// ===========================================================================
extern "C" void kernel_launch(void* const* d_in, const int* in_sizes, int n_in,
                              void* d_out, int out_size, void* d_ws, size_t ws_size,
                              hipStream_t stream) {
    (void)in_sizes; (void)n_in; (void)out_size;
    const float* x      = (const float*)d_in[0];
    const float* w      = (const float*)d_in[1];
    const float* scales = (const float*)d_in[2];
    float* out          = (float*)d_out;

    const size_t w_elems = (size_t)NOUT * KDIM;     // 117M
    const size_t x_elems = (size_t)TOKENS * KDIM;   // 8.4M
    const size_t need    = (w_elems + x_elems) * sizeof(_Float16);  // ~251 MB

    dim3 grid(TOKENS / M_TILE, NOUT / N_TILE);  // (16, 224), M-fastest

    if (d_ws != nullptr && ws_size >= need) {
        _Float16* wq = (_Float16*)d_ws;
        _Float16* xh = wq + w_elems;
        dequant_w_kernel<<<(unsigned)(w_elems / 2048), 256, 0, stream>>>(w, scales, wq);
        cvt_x_kernel<<<(unsigned)(x_elems / 2048), 256, 0, stream>>>(x, xh);
        gemm_f16_async_kernel<<<grid, 256, 0, stream>>>(xh, wq, out);
    } else {
        gql_wmma_fused_kernel<<<grid, 256, 0, stream>>>(x, w, scales, out);
    }
}
